// Partition_layer_64665027609000
// MI455X (gfx1250) — matrix operand order repaired and checked by reading, not compiled, not run on hardware
//
#include <hip/hip_runtime.h>

typedef unsigned short u16;
typedef unsigned int   u32;
typedef __attribute__((ext_vector_type(16))) __bf16 v16bf;
typedef __attribute__((ext_vector_type(8)))  float  v8f;

#define Jn   22
#define Nn   1408
#define Cc   256
#define Bb   32
#define NNe  ((size_t)Nn * Nn)        // 1982464
#define ROWS (Bb * Nn)                // 45056
#define BNC  ((size_t)ROWS * Cc)      // 11534336
#define EPSV 1e-5f
#define NEGV -9e15f

struct __attribute__((aligned(16))) U4 { u32 x, y, z, w; };
struct Q2 { U4 a, b; };

__device__ __forceinline__ u16 f2bf(float f) {
    u32 u = __builtin_bit_cast(u32, f);
    u32 r = u + 0x7FFFu + ((u >> 16) & 1u);
    return (u16)(r >> 16);
}

__device__ __forceinline__ v16bf frag_from(const u16* p0, const u16* p1) {
    Q2 q;
    q.a = *(const U4*)p0;
    q.b = *(const U4*)p1;
    return __builtin_bit_cast(v16bf, q);
}

// CDNA5 async global->LDS copy (ASYNCcnt-tracked). lds = LDS byte offset,
// g = per-lane 64-bit global address. GV mode: vdst, vaddr, off.
__device__ __forceinline__ void async_b128(u32 lds, const void* g) {
    asm volatile("global_load_async_to_lds_b128 %0, %1, off"
                 :: "v"(lds), "v"(g) : "memory");
}

// ---------------- elementwise f32 -> bf16 ----------------
__global__ void cvt_f32_bf16(const float* __restrict__ in, u16* __restrict__ out, size_t n) {
    size_t i = (size_t)blockIdx.x * blockDim.x + threadIdx.x;
    size_t stride = (size_t)gridDim.x * blockDim.x;
    for (; i < n; i += stride) out[i] = f2bf(in[i]);
}

// ---------------- 256x256 f32 -> bf16 transpose (weights) ----------------
__global__ void cvt_transpose_bf16(const float* __restrict__ in, u16* __restrict__ out) {
    int idx = blockIdx.x * blockDim.x + threadIdx.x;   // 65536 threads total
    int i = idx >> 8, j = idx & 255;
    out[(size_t)j * Cc + i] = f2bf(in[idx]);
}

// ---------------- masked softmax adjacency: s_adj, I_s (bf16) ----------------
__global__ void softmax_adj(const float* __restrict__ A, const float* __restrict__ I,
                            const float* __restrict__ edg,
                            u16* __restrict__ sadj, u16* __restrict__ isadj) {
    __shared__ float se[Jn * Jn];
    for (int t = threadIdx.x; t < Jn * Jn; t += blockDim.x) se[t] = edg[t];
    __syncthreads();
    int wave = threadIdx.x >> 5, lane = threadIdx.x & 31;
    int i = blockIdx.x * 8 + wave;            // 176 blocks * 8 waves = 1408 rows
    if (i >= Nn) return;
    int ib = i / Jn, ij = i - ib * Jn;
    const float* Ar = A + (size_t)i * Nn;
    const float* Ir = I + (size_t)i * Nn;
    float mx = NEGV;
    for (int j = lane; j < Nn; j += 32) {
        float comb = Ar[j] + Ir[j];
        float bd = ((j / Jn) == ib) ? se[ij * Jn + (j % Jn)] : 0.f;
        float sc = (comb > 0.f) ? bd : NEGV;
        mx = fmaxf(mx, sc);
    }
    for (int o = 16; o; o >>= 1) mx = fmaxf(mx, __shfl_xor(mx, o, 32));
    float sum = 0.f;
    for (int j = lane; j < Nn; j += 32) {
        float comb = Ar[j] + Ir[j];
        float bd = ((j / Jn) == ib) ? se[ij * Jn + (j % Jn)] : 0.f;
        float sc = (comb > 0.f) ? bd : NEGV;
        sum += __expf(sc - mx);
    }
    for (int o = 16; o; o >>= 1) sum += __shfl_xor(sum, o, 32);
    float inv = 1.f / sum;
    for (int j = lane; j < Nn; j += 32) {
        float a = Ar[j], ii = Ir[j];
        float comb = a + ii;
        float bd = ((j / Jn) == ib) ? se[ij * Jn + (j % Jn)] : 0.f;
        float sc = (comb > 0.f) ? bd : NEGV;
        float sm = __expf(sc - mx) * inv;
        sadj[(size_t)i * Nn + j]  = f2bf(a * sm);
        isadj[(size_t)i * Nn + j] = f2bf(ii * sm);
    }
}

// ---------------- bias combine ----------------
__global__ void combine_bias(const float* b0, const float* b1, const float* b2, float* out) {
    int c = threadIdx.x;
    float v = b0[c] + b1[c];
    if (b2) v += b2[c];
    out[c] = v;
}

// ---------------- WMMA GEMM: Out = sum_p A_p[M,K] @ Bt_p^T (+bias) ----------------
// A row-major [M,K] (ldA); B supplied K-major ("B^T", [Nc,K], ldBT).
// bf16 inputs, f32 accumulation via v_wmma_f32_16x16x32_bf16.
// Double-buffered LDS staged with global_load_async_to_lds_b128 + s_wait_asynccnt.
// The p (term) loop is compile-time unrolled so all base pointers stay in SGPRs
// (no dynamic pointer arrays -> no scratch spills).
// OUT_BF16_T: bf16 output written transposed [Nc, M] (ldOut = M), vector b128 stores.
// else: f32 output row-major [M, Nc] (ldOut = Nc) + per-column bias.
template<int P, bool OUT_BF16_T>
__global__ __launch_bounds__(256)
void wmma_gemm(const u16* __restrict__ A0, const u16* __restrict__ A1, const u16* __restrict__ A2,
               const u16* __restrict__ B0, const u16* __restrict__ B1, const u16* __restrict__ B2,
               void* __restrict__ Out, const float* __restrict__ bias,
               int K, int ldA, int ldBT, int ldOut,
               long long sAb, long long sBb, long long sOb) {
    __shared__ u16 At[2][128][40];   // [stage][m][k], padded rows (80B, 16B-aligned)
    __shared__ u16 Bt[2][128][40];   // [stage][n][k]
    const int tid  = threadIdx.x;
    const int lane = tid & 31, wave = tid >> 5;
    const int wm = wave & 3, wn = wave >> 2;
    const int hl = lane >> 4, lrow = lane & 15;
    const int r = tid >> 1, seg = tid & 1;      // staging coords: 128 rows x 2 segs
    const size_t blockM = (size_t)blockIdx.x * 128;
    const size_t blockN = (size_t)blockIdx.y * 128;
    const int batch = blockIdx.z;

    const size_t aoff = (size_t)batch * sAb + (blockM + r) * (size_t)ldA  + seg * 16;
    const size_t boff = (size_t)batch * sBb + (blockN + r) * (size_t)ldBT + seg * 16;
    const u32 STG = (u32)sizeof(At[0]);          // bytes per stage (10240)
    const u32 la0 = (u32)(size_t)&At[0][r][seg * 16];
    const u32 lb0 = (u32)(size_t)&Bt[0][r][seg * 16];

    v8f zero = {0.f, 0.f, 0.f, 0.f, 0.f, 0.f, 0.f, 0.f};
    v8f acc[2][4];
#pragma unroll
    for (int i = 0; i < 2; ++i)
#pragma unroll
        for (int j = 0; j < 4; ++j) acc[i][j] = zero;

    auto prefetch = [&](int st, const u16* Aq, const u16* Bq, int kc) {
        const u16* Ag = Aq + aoff + kc;
        const u16* Bg = Bq + boff + kc;
        u32 la = la0 + (u32)st * STG;
        u32 lb = lb0 + (u32)st * STG;
        async_b128(la,      Ag);
        async_b128(la + 16, Ag + 8);
        async_b128(lb,      Bg);
        async_b128(lb + 16, Bg + 8);
    };

    auto compute = [&](int st) {
        v16bf af[2], bfr[4];
#pragma unroll
        for (int ti = 0; ti < 2; ++ti) {
            int rb = wm * 32 + ti * 16 + lrow;
            // A 16x32 bf16: lanes0-15 K=0..7 & 16..23; lanes16-31 K=8..15 & 24..31
            af[ti] = frag_from(&At[st][rb][hl * 8], &At[st][rb][16 + hl * 8]);
        }
#pragma unroll
        for (int tj = 0; tj < 4; ++tj) {
            int nb = wn * 64 + tj * 16 + lrow;
            // B 32x16 bf16: lanes0-15 K=0..7 & 16..23; lanes16-31 K=8..15 & 24..31
            bfr[tj] = frag_from(&Bt[st][nb][hl * 8], &Bt[st][nb][16 + hl * 8]);
        }
#pragma unroll
        for (int ti = 0; ti < 2; ++ti)
#pragma unroll
            for (int tj = 0; tj < 4; ++tj)
                acc[ti][tj] = __builtin_amdgcn_wmma_f32_16x16x32_bf16(
                    false, af[ti], false, bfr[tj], (short)0, acc[ti][tj], false, false);
    };

    int stage = 0;
    prefetch(0, A0, B0, 0);                     // prologue prefetch into stage 0
#pragma unroll
    for (int p = 0; p < P; ++p) {
        const u16* Ac = (p == 0) ? A0 : ((p == 1) ? A1 : A2);   // compile-time select
        const u16* Bc = (p == 0) ? B0 : ((p == 1) ? B1 : B2);
        const u16* An = (p == 0) ? A1 : A2;     // next-term base (used iff p+1 < P)
        const u16* Bn = (p == 0) ? B1 : B2;
        for (int kc = 0; kc < K; kc += 32) {
            const bool lastAll = (p == P - 1) && (kc + 32 >= K);
            if (!lastAll) {
                if (kc + 32 < K) prefetch(stage ^ 1, Ac, Bc, kc + 32);
                else             prefetch(stage ^ 1, An, Bn, 0);
                asm volatile("s_wait_asynccnt 0x4" ::: "memory");  // current stage done
            } else {
                asm volatile("s_wait_asynccnt 0x0" ::: "memory");
            }
            __syncthreads();
            compute(stage);
            __syncthreads();
            stage ^= 1;
        }
    }

    const size_t outBase = (size_t)batch * (size_t)sOb;
#pragma unroll
    for (int ti = 0; ti < 2; ++ti) {
#pragma unroll
        for (int tj = 0; tj < 4; ++tj) {
            size_t gn  = blockN + wn * 64 + tj * 16 + lrow;
            size_t gm0 = blockM + wm * 32 + ti * 16 + hl * 8;   // 8 contiguous M rows
            if constexpr (OUT_BF16_T) {
                u16 __attribute__((aligned(16))) tmp[8];
#pragma unroll
                for (int q = 0; q < 8; ++q) tmp[q] = f2bf(acc[ti][tj][q]);
                *(U4*)((u16*)Out + outBase + gn * (size_t)ldOut + gm0) = *(const U4*)tmp;
            } else {
                float bv = bias ? bias[gn] : 0.f;
#pragma unroll
                for (int q = 0; q < 8; ++q)
                    ((float*)Out)[outBase + (gm0 + q) * (size_t)ldOut + gn] = acc[ti][tj][q] + bv;
            }
        }
    }
}

// ---------------- BatchNorm stats / apply ----------------
__global__ void bn_zero(float* stats) { stats[threadIdx.x] = 0.f; }

__global__ void bn_stats(const float* __restrict__ in, float* __restrict__ stats) {
    int c = threadIdx.x;   // 256 channels
    float s = 0.f, s2 = 0.f;
    for (int r = blockIdx.x; r < ROWS; r += gridDim.x) {
        float v = in[(size_t)r * Cc + c];
        s += v; s2 += v * v;
    }
    atomicAdd(&stats[c], s);
    atomicAdd(&stats[Cc + c], s2);
}

__global__ void bn_relu_bf16(const float* __restrict__ in, const float* __restrict__ stats,
                             const float* __restrict__ g, const float* __restrict__ b,
                             u16* __restrict__ out) {
    size_t i = (size_t)blockIdx.x * blockDim.x + threadIdx.x;
    size_t stride = (size_t)gridDim.x * blockDim.x;
    const float invR = 1.f / (float)ROWS;
    for (; i < BNC; i += stride) {
        int c = (int)(i & (Cc - 1));
        float mean = stats[c] * invR;
        float var  = stats[Cc + c] * invR - mean * mean;
        float y = (in[i] - mean) * rsqrtf(var + EPSV) * g[c] + b[c];
        out[i] = f2bf(fmaxf(y, 0.f));
    }
}

__global__ void bn_res_relu(const float* __restrict__ in, const float* __restrict__ stats,
                            const float* __restrict__ g, const float* __restrict__ b,
                            const float* __restrict__ xres, float* __restrict__ out) {
    size_t i = (size_t)blockIdx.x * blockDim.x + threadIdx.x;
    size_t stride = (size_t)gridDim.x * blockDim.x;
    const float invR = 1.f / (float)ROWS;
    for (; i < BNC; i += stride) {
        int c = (int)(i & (Cc - 1));
        float mean = stats[c] * invR;
        float var  = stats[Cc + c] * invR - mean * mean;
        float y = (in[i] - mean) * rsqrtf(var + EPSV) * g[c] + b[c];
        out[i] = fmaxf(y + xres[i], 0.f);
    }
}

// ---------------- host launch ----------------
extern "C" void kernel_launch(void* const* d_in, const int* in_sizes, int n_in,
                              void* d_out, int out_size, void* d_ws, size_t ws_size,
                              hipStream_t stream) {
    const float* x      = (const float*)d_in[0];
    const float* adj    = (const float*)d_in[1];
    const float* edg    = (const float*)d_in[2];
    const float* w_s    = (const float*)d_in[3];
    const float* b_s    = (const float*)d_in[4];
    const float* w_is   = (const float*)d_in[5];
    const float* b_is   = (const float*)d_in[6];
    const float* w_prev = (const float*)d_in[7];
    const float* b_prev = (const float*)d_in[8];
    const float* w_next = (const float*)d_in[9];
    const float* b_next = (const float*)d_in[10];
    const float* w_it   = (const float*)d_in[11];
    const float* b_it   = (const float*)d_in[12];
    const float* bn_s_w = (const float*)d_in[13];
    const float* bn_s_b = (const float*)d_in[14];
    const float* bn_t_w = (const float*)d_in[15];
    const float* bn_t_b = (const float*)d_in[16];

    const float* A_sp  = adj;
    const float* prevA = adj + NNe;
    const float* I_sp  = adj + 3 * NNe;

    char* ws = (char*)d_ws;
    size_t off = 0;
    auto alloc = [&](size_t bytes) -> char* {
        char* p = ws + off;
        off = (off + bytes + 255) & ~(size_t)255;
        return p;
    };
    const size_t CCn = (size_t)Cc * Cc;
    u16*   sadj  = (u16*)alloc(NNe * 2);
    u16*   isadj = (u16*)alloc(NNe * 2);
    u16*   pni   = (u16*)alloc(3 * NNe * 2);      // prev, next, I (bf16)
    u16*   xbf   = (u16*)alloc(BNC * 2);          // x row-major bf16
    u16*   wbfT  = (u16*)alloc(5 * CCn * 2);      // W^T: s, is, prev, next, it
    u16*   xw0T  = (u16*)alloc(BNC * 2);          // (x@w_s)^T  [C, ROWS]
    u16*   xw1T  = (u16*)alloc(BNC * 2);          // (x@w_is)^T
    float* hpre  = (float*)alloc(BNC * 4);        // reused for h2
    u16*   hbf   = (u16*)alloc(BNC * 2);          // h row-major bf16
    u16*   hwT   = (u16*)alloc(3 * BNC * 2);      // (h@w_*)^T: it, prev, next
    float* stats = (float*)alloc(2 * Cc * 4);
    float* bias0 = (float*)alloc(Cc * 4);
    float* bias1 = (float*)alloc(Cc * 4);
    if (off > ws_size) return;

    u16* prevbf = pni;
    u16* nextbf = pni + NNe;
    u16* Ibf    = pni + 2 * NNe;
    u16* wT_s = wbfT, *wT_is = wbfT + CCn, *wT_prev = wbfT + 2 * CCn,
       * wT_next = wbfT + 3 * CCn, *wT_it = wbfT + 4 * CCn;
    u16* hwT_it = hwT, *hwT_prev = hwT + BNC, *hwT_next = hwT + 2 * BNC;

    // Phase 0: conversions + adjacency softmax + biases
    cvt_f32_bf16<<<4096, 256, 0, stream>>>(x, xbf, BNC);
    cvt_f32_bf16<<<2048, 256, 0, stream>>>(prevA, pni, 3 * NNe);
    cvt_transpose_bf16<<<256, 256, 0, stream>>>(w_s, wT_s);
    cvt_transpose_bf16<<<256, 256, 0, stream>>>(w_is, wT_is);
    cvt_transpose_bf16<<<256, 256, 0, stream>>>(w_prev, wT_prev);
    cvt_transpose_bf16<<<256, 256, 0, stream>>>(w_next, wT_next);
    cvt_transpose_bf16<<<256, 256, 0, stream>>>(w_it, wT_it);
    softmax_adj<<<Nn / 8, 256, 0, stream>>>(A_sp, I_sp, edg, sadj, isadj);
    combine_bias<<<1, Cc, 0, stream>>>(b_s, b_is, nullptr, bias0);
    combine_bias<<<1, Cc, 0, stream>>>(b_it, b_prev, b_next, bias1);

    dim3 gW(ROWS / 128, Cc / 128, 1);     // (352, 2, 1)
    dim3 gA(Nn / 128, Cc / 128, Bb);      // (11, 2, 32)
    const long long NC = (long long)Nn * Cc;

    // Phase 1: xw_*^T = (x @ w_*)^T  (bf16, transposed vector stores)
    wmma_gemm<1, true><<<gW, 256, 0, stream>>>(xbf, nullptr, nullptr, wT_s, nullptr, nullptr,
                                               xw0T, nullptr, Cc, Cc, Cc, ROWS, 0, 0, 0);
    wmma_gemm<1, true><<<gW, 256, 0, stream>>>(xbf, nullptr, nullptr, wT_is, nullptr, nullptr,
                                               xw1T, nullptr, Cc, Cc, Cc, ROWS, 0, 0, 0);
    // hpre = s_adj@xw_s + I_s@xw_is + (b_s+b_is), batched over 32
    // B^T layout [C, ROWS]: per-batch K-offset = batch*Nn
    wmma_gemm<2, false><<<gA, 256, 0, stream>>>(sadj, isadj, nullptr, xw0T, xw1T, nullptr,
                                                hpre, bias0, Nn, Nn, ROWS, Cc, 0, Nn, NC);
    // BN + ReLU -> h (bf16 row-major)
    bn_zero<<<1, 2 * Cc, 0, stream>>>(stats);
    bn_stats<<<256, Cc, 0, stream>>>(hpre, stats);
    bn_relu_bf16<<<4096, 256, 0, stream>>>(hpre, stats, bn_s_w, bn_s_b, hbf);

    // Phase 2: hw_*^T = (h @ w_*)^T
    wmma_gemm<1, true><<<gW, 256, 0, stream>>>(hbf, nullptr, nullptr, wT_it, nullptr, nullptr,
                                               hwT_it, nullptr, Cc, Cc, Cc, ROWS, 0, 0, 0);
    wmma_gemm<1, true><<<gW, 256, 0, stream>>>(hbf, nullptr, nullptr, wT_prev, nullptr, nullptr,
                                               hwT_prev, nullptr, Cc, Cc, Cc, ROWS, 0, 0, 0);
    wmma_gemm<1, true><<<gW, 256, 0, stream>>>(hbf, nullptr, nullptr, wT_next, nullptr, nullptr,
                                               hwT_next, nullptr, Cc, Cc, Cc, ROWS, 0, 0, 0);
    // h2 = I@hw_it + prev@hw_prev + next@hw_next + (b_it+b_prev+b_next)  (reuse hpre)
    wmma_gemm<3, false><<<gA, 256, 0, stream>>>(Ibf, prevbf, nextbf, hwT_it, hwT_prev, hwT_next,
                                                hpre, bias1, Nn, Nn, ROWS, Cc, 0, Nn, NC);
    // BN + residual + ReLU -> out (f32)
    bn_zero<<<1, 2 * Cc, 0, stream>>>(stats);
    bn_stats<<<256, Cc, 0, stream>>>(hpre, stats);
    bn_res_relu<<<4096, 256, 0, stream>>>(hpre, stats, bn_t_w, bn_t_b, x, (float*)d_out);
}
